// FlowformerLayer_111669150276
// MI455X (gfx1250) — matrix operand order repaired and checked by reading, not compiled, not run on hardware
//
#include <hip/hip_runtime.h>
#include <hip/hip_bf16.h>
#include <cstdint>

typedef __bf16 bf16_t;
typedef __attribute__((ext_vector_type(16))) __bf16 v16bf;
typedef __attribute__((ext_vector_type(8)))  float  v8f;

#define L_SEQ   16384
#define D_MODEL 512
#define F_FFN   2048
#define N_HEADS 8
#define D_HEAD  64
#define EPS_F   1e-6f

__device__ __forceinline__ bf16_t f2bf(float f) {
  union { float f; uint32_t u; } x; x.f = f;
  uint32_t r = (x.u + 0x7FFFu + ((x.u >> 16) & 1u)) >> 16;
  union { uint16_t s; bf16_t b; } y; y.s = (uint16_t)r;
  return y.b;
}

// LDS byte offset of a pointer into a __shared__ array (ptrtoint of AS(3)).
template <typename T>
__device__ __forceinline__ uint32_t lds_off(T* p) {
  return (uint32_t)(uintptr_t)((__attribute__((address_space(3))) T*)p);
}

// CDNA5 async global -> LDS copy, 16B per lane, tracked by ASYNCcnt.
__device__ __forceinline__ void async_copy_b128(uint32_t lds_byte_off, const void* gaddr) {
  asm volatile("global_load_async_to_lds_b128 %0, %1, off"
               :: "v"(lds_byte_off), "v"((uint64_t)(uintptr_t)gaddr)
               : "memory");
}

__device__ __forceinline__ void wait_async0() {
  asm volatile("s_wait_asynccnt 0x0" ::: "memory");
}

// ---------------------------------------------------------------------------
// f32 -> bf16 elementwise (weight casting)
// ---------------------------------------------------------------------------
__global__ __launch_bounds__(256) void cvt_bf16_kernel(const float* __restrict__ in,
                                                       bf16_t* __restrict__ out, int n) {
  int i = blockIdx.x * 256 + threadIdx.x;
  if (i < n) out[i] = f2bf(in[i]);
}

// ---------------------------------------------------------------------------
// LayerNorm over D=512, one block per row, bf16 output
// ---------------------------------------------------------------------------
__global__ __launch_bounds__(256) void layernorm_kernel(const float* __restrict__ x,
                                                        const float* __restrict__ g,
                                                        const float* __restrict__ b,
                                                        bf16_t* __restrict__ out) {
  const int row = blockIdx.x;
  const int tid = threadIdx.x;
  const float* xr = x + (size_t)row * D_MODEL;
  float v0 = xr[tid], v1 = xr[tid + 256];
  __shared__ float red[256];
  __shared__ float stat;
  red[tid] = v0 + v1;
  __syncthreads();
  for (int o = 128; o > 0; o >>= 1) { if (tid < o) red[tid] += red[tid + o]; __syncthreads(); }
  if (tid == 0) stat = red[0] * (1.0f / D_MODEL);
  __syncthreads();
  float mean = stat;
  float d0 = v0 - mean, d1 = v1 - mean;
  __syncthreads();
  red[tid] = d0 * d0 + d1 * d1;
  __syncthreads();
  for (int o = 128; o > 0; o >>= 1) { if (tid < o) red[tid] += red[tid + o]; __syncthreads(); }
  if (tid == 0) stat = red[0] * (1.0f / D_MODEL);
  __syncthreads();
  float inv = rsqrtf(stat + 1e-5f);
  bf16_t* orow = out + (size_t)row * D_MODEL;
  orow[tid]       = f2bf(d0 * inv * g[tid]       + b[tid]);
  orow[tid + 256] = f2bf(d1 * inv * g[tid + 256] + b[tid + 256]);
}

// ---------------------------------------------------------------------------
// WMMA bf16 GEMM: C[MxN] = A[MxK] * B[KxN] (+bias) with fused epilogues.
// Block tile 128x128, K-step 32, 256 threads = 8 waves (4x2), wave tile 32x64.
// Double-buffered async (GLOBAL_LOAD_ASYNC_TO_LDS_B128) staging pipeline.
// MODE: 0 plain->f32, 1 sigmoid->f32, 2 gelu->bf16, 3 plain+residual->f32
// ---------------------------------------------------------------------------
#define BM 128
#define BN 128
#define BK 32

union Frag { v16bf v; uint4 u[2]; };

template <int MODE>
__global__ __launch_bounds__(256)
void gemm_bf16_kernel(const bf16_t* __restrict__ A, const bf16_t* __restrict__ Bw,
                      const float* __restrict__ bias, const float* __restrict__ residual,
                      float* __restrict__ Cf, bf16_t* __restrict__ Cb,
                      int M, int N, int K) {
  __shared__ __align__(16) bf16_t As[2][BM][BK + 8];   // row stride 40 elems (80B)
  __shared__ __align__(16) bf16_t Bs[2][BK][BN + 8];   // row stride 136 elems (272B)

  const int tid  = threadIdx.x;
  const int lane = tid & 31;
  const int wave = tid >> 5;
  const int wr   = wave & 3;       // wave row (M dir), 4 waves
  const int wc   = wave >> 2;      // wave col (N dir), 2 waves
  const int bm   = blockIdx.y * BM;
  const int bn   = blockIdx.x * BN;
  const int lm   = lane & 15;      // A fragment: M within tile
  const int lk   = lane >> 4;      // A fragment: K-half select

  // per-thread staging coordinates (A: 512 uint4; B: 512 uint4; 2 each)
  const int ar0 = tid >> 2,          aq0 = tid & 3;
  const int ar1 = (tid + 256) >> 2,  aq1 = (tid + 256) & 3;
  const int br0 = tid >> 4,          bq0 = tid & 15;
  const int br1 = (tid + 256) >> 4,  bq1 = (tid + 256) & 15;

  v8f acc[2][4];
#pragma unroll
  for (int im = 0; im < 2; ++im)
#pragma unroll
    for (int in = 0; in < 4; ++in) {
      v8f z = {0.f, 0.f, 0.f, 0.f, 0.f, 0.f, 0.f, 0.f};
      acc[im][in] = z;
    }

  auto stage = [&](int buf, int k0) {
    async_copy_b128(lds_off(&As[buf][ar0][aq0 * 8]), A + (size_t)(bm + ar0) * K + k0 + aq0 * 8);
    async_copy_b128(lds_off(&As[buf][ar1][aq1 * 8]), A + (size_t)(bm + ar1) * K + k0 + aq1 * 8);
    async_copy_b128(lds_off(&Bs[buf][br0][bq0 * 8]), Bw + (size_t)(k0 + br0) * N + bn + bq0 * 8);
    async_copy_b128(lds_off(&Bs[buf][br1][bq1 * 8]), Bw + (size_t)(k0 + br1) * N + bn + bq1 * 8);
  };

  const int ktiles = K / BK;
  stage(0, 0);
  wait_async0();
  __syncthreads();

  for (int t = 0; t < ktiles; ++t) {
    const int cur = t & 1;
    if (t + 1 < ktiles) stage(cur ^ 1, (t + 1) * BK);   // DMA next tile in background

    // --- load fragments per ISA wave32 layouts ---
    Frag af[2];
#pragma unroll
    for (int im = 0; im < 2; ++im) {
      int m = wr * 32 + im * 16 + lm;
      af[im].u[0] = *(const uint4*)(&As[cur][m][lk * 8]);        // K 0..7  / 8..15
      af[im].u[1] = *(const uint4*)(&As[cur][m][16 + lk * 8]);   // K 16..23 / 24..31
    }
    Frag bfr[4];
#pragma unroll
    for (int in = 0; in < 4; ++in) {
      int n = wc * 64 + in * 16;
#pragma unroll
      for (int s = 0; s < 16; ++s)
        bfr[in].v[s] = Bs[cur][(s < 8) ? (lk * 8 + s) : (16 + lk * 8 + (s - 8))][n + lm];
    }

#pragma unroll
    for (int im = 0; im < 2; ++im)
#pragma unroll
      for (int in = 0; in < 4; ++in)
        acc[im][in] = __builtin_amdgcn_wmma_f32_16x16x32_bf16(
            false, af[im].v, false, bfr[in].v, (short)0, acc[im][in], false, false);

    if (t + 1 < ktiles) {
      wait_async0();        // own async copies for next buffer landed
      __syncthreads();      // whole tile resident; nobody still reads old buffer
    }
  }

  // --- epilogue: C/D layout: VGPR j -> M = j + 8*(lane>=16), N = lane%16 ---
  const int mh = 8 * lk;
#pragma unroll
  for (int im = 0; im < 2; ++im)
#pragma unroll
    for (int in = 0; in < 4; ++in)
#pragma unroll
      for (int j = 0; j < 8; ++j) {
        int gm = bm + wr * 32 + im * 16 + j + mh;
        int gn = bn + wc * 64 + in * 16 + lm;
        float val = acc[im][in][j] + bias[gn];
        size_t idx = (size_t)gm * N + gn;
        if (MODE == 0) {
          Cf[idx] = val;
        } else if (MODE == 1) {
          Cf[idx] = 1.0f / (1.0f + expf(-val));
        } else if (MODE == 2) {
          float ge = 0.5f * val * (1.0f + erff(val * 0.70710678118654752f));
          Cb[idx] = f2bf(ge);
        } else {
          Cf[idx] = val + residual[idx];
        }
      }
}

// ---------------------------------------------------------------------------
// Column sums over L: qsum[c] = sum_l q[l][c], ksum[c] = sum_l k[l][c]
// ---------------------------------------------------------------------------
__global__ __launch_bounds__(256) void colsum2_kernel(const float* __restrict__ q,
                                                      const float* __restrict__ k,
                                                      float* __restrict__ qs,
                                                      float* __restrict__ ks) {
  int c = blockIdx.x, tid = threadIdx.x;
  float sq = 0.f, sk = 0.f;
  for (int r = tid; r < L_SEQ; r += 256) {
    sq += q[(size_t)r * D_MODEL + c];
    sk += k[(size_t)r * D_MODEL + c];
  }
  __shared__ float r1[256], r2[256];
  r1[tid] = sq; r2[tid] = sk;
  __syncthreads();
  for (int o = 128; o > 0; o >>= 1) { if (tid < o) { r1[tid] += r1[tid+o]; r2[tid] += r2[tid+o]; } __syncthreads(); }
  if (tid == 0) { qs[c] = r1[0]; ks[c] = r2[0]; }
}

// Weighted column sums: qw[c]=sum_l q*sink_in[h(c)][l], kw[c]=sum_l k*src_out[h(c)][l]
__global__ __launch_bounds__(256) void wcolsum_kernel(const float* __restrict__ q,
                                                      const float* __restrict__ k,
                                                      const float* __restrict__ sink_in,
                                                      const float* __restrict__ src_out,
                                                      float* __restrict__ qw,
                                                      float* __restrict__ kw) {
  int c = blockIdx.x, tid = threadIdx.x;
  int h = c >> 6;
  float sq = 0.f, sk = 0.f;
  for (int r = tid; r < L_SEQ; r += 256) {
    sq += q[(size_t)r * D_MODEL + c] * sink_in[h * L_SEQ + r];
    sk += k[(size_t)r * D_MODEL + c] * src_out[h * L_SEQ + r];
  }
  __shared__ float r1[256], r2[256];
  r1[tid] = sq; r2[tid] = sk;
  __syncthreads();
  for (int o = 128; o > 0; o >>= 1) { if (tid < o) { r1[tid] += r1[tid+o]; r2[tid] += r2[tid+o]; } __syncthreads(); }
  if (tid == 0) { qw[c] = r1[0]; kw[c] = r2[0]; }
}

// ---------------------------------------------------------------------------
// Per-(h,l) dots: sink_incoming / source_outgoing (one wave32 per pair)
// ---------------------------------------------------------------------------
__global__ __launch_bounds__(256) void flow_inv_kernel(const float* __restrict__ q,
                                                       const float* __restrict__ k,
                                                       const float* __restrict__ ksum,
                                                       const float* __restrict__ qsum,
                                                       float* __restrict__ sink_in,
                                                       float* __restrict__ src_out) {
  int gw = (blockIdx.x * 256 + threadIdx.x) >> 5;
  int lane = threadIdx.x & 31;
  int nwaves = gridDim.x * 8;
  for (int p = gw; p < N_HEADS * L_SEQ; p += nwaves) {
    int h = p >> 14, l = p & (L_SEQ - 1);
    size_t base = (size_t)l * D_MODEL + h * D_HEAD;
    float s1 = 0.f, s2 = 0.f;
    for (int d = lane; d < D_HEAD; d += 32) {
      s1 += (q[base + d] + EPS_F) * (ksum[h * D_HEAD + d] + EPS_F);
      s2 += (k[base + d] + EPS_F) * (qsum[h * D_HEAD + d] + EPS_F);
    }
    for (int o = 16; o > 0; o >>= 1) { s1 += __shfl_xor(s1, o); s2 += __shfl_xor(s2, o); }
    if (lane == 0) { sink_in[p] = 1.0f / s1; src_out[p] = 1.0f / s2; }
  }
}

// conserved_sink -> sink_alloc = sigmoid(cs); conserved_source clipped to [-1,1]
__global__ __launch_bounds__(256) void flow_cons_kernel(const float* __restrict__ q,
                                                        const float* __restrict__ k,
                                                        const float* __restrict__ kw,
                                                        const float* __restrict__ qw,
                                                        float* __restrict__ sink_alloc,
                                                        float* __restrict__ cons_src) {
  int gw = (blockIdx.x * 256 + threadIdx.x) >> 5;
  int lane = threadIdx.x & 31;
  int nwaves = gridDim.x * 8;
  for (int p = gw; p < N_HEADS * L_SEQ; p += nwaves) {
    int h = p >> 14, l = p & (L_SEQ - 1);
    size_t base = (size_t)l * D_MODEL + h * D_HEAD;
    float s1 = 0.f, s2 = 0.f;
    for (int d = lane; d < D_HEAD; d += 32) {
      s1 += (q[base + d] + EPS_F) * (kw[h * D_HEAD + d] + EPS_F);
      s2 += (k[base + d] + EPS_F) * (qw[h * D_HEAD + d] + EPS_F);
    }
    for (int o = 16; o > 0; o >>= 1) { s1 += __shfl_xor(s1, o); s2 += __shfl_xor(s2, o); }
    if (lane == 0) {
      sink_alloc[p] = 1.0f / (1.0f + expf(-s1));   // L/S == 1
      cons_src[p]   = fminf(fmaxf(s2, -1.0f), 1.0f);
    }
  }
}

// Softmax over L per head (deterministic two-stage). cons_src in [-1,1].
__global__ __launch_bounds__(256) void softmax_sum_kernel(const float* __restrict__ cs,
                                                          float* __restrict__ part) {
  int blk = blockIdx.x, tid = threadIdx.x;      // 128 blocks: h = blk/16, seg = blk%16
  int h = blk >> 4, seg = blk & 15;
  float s = 0.f;
  for (int i = tid; i < 1024; i += 256) s += expf(cs[h * L_SEQ + seg * 1024 + i]);
  __shared__ float red[256];
  red[tid] = s; __syncthreads();
  for (int o = 128; o > 0; o >>= 1) { if (tid < o) red[tid] += red[tid + o]; __syncthreads(); }
  if (tid == 0) part[blk] = red[0];
}

__global__ __launch_bounds__(256) void softmax_comp_kernel(const float* __restrict__ cs,
                                                           const float* __restrict__ part,
                                                           float* __restrict__ comp) {
  int p = blockIdx.x * 256 + threadIdx.x;
  int h = p >> 14;
  float denom = 0.f;
#pragma unroll
  for (int s = 0; s < 16; ++s) denom += part[h * 16 + s];
  comp[p] = expf(cs[p]) * ((float)L_SEQ) / denom;
}

// ---------------------------------------------------------------------------
// kv[h][d][m] = sum_l k[l][h,d] * v[l][h,m] * comp[h][l]   (split-L partials)
// ---------------------------------------------------------------------------
#define KV_SPLITS 32
__global__ __launch_bounds__(256) void kv_partial_kernel(const float* __restrict__ k,
                                                         const float* __restrict__ v,
                                                         const float* __restrict__ comp,
                                                         float* __restrict__ part) {
  const int h = blockIdx.y, sp = blockIdx.x, t = threadIdx.x;
  const int l0 = sp * (L_SEQ / KV_SPLITS);     // 512 rows per split
  const int d = t >> 2, mq = t & 3;
  __shared__ float ks[8][64], vs[8][64];
  float acc[16];
#pragma unroll
  for (int j = 0; j < 16; ++j) acc[j] = 0.f;

  for (int ch = 0; ch < 512; ch += 8) {
    __syncthreads();
#pragma unroll
    for (int it = 0; it < 2; ++it) {
      int i = t + it * 256;
      int r = i >> 6, c = i & 63;
      int l = l0 + ch + r;
      ks[r][c] = k[(size_t)l * D_MODEL + h * D_HEAD + c];
      vs[r][c] = v[(size_t)l * D_MODEL + h * D_HEAD + c] * comp[h * L_SEQ + l];
    }
    __syncthreads();
#pragma unroll
    for (int r = 0; r < 8; ++r) {
      float kd = ks[r][d];
#pragma unroll
      for (int j = 0; j < 16; ++j) acc[j] = fmaf(kd, vs[r][mq * 16 + j], acc[j]);
    }
  }
  float* outp = part + (((size_t)h * KV_SPLITS + sp) * 64 + d) * 64 + mq * 16;
#pragma unroll
  for (int j = 0; j < 16; ++j) outp[j] = acc[j];
}

__global__ __launch_bounds__(256) void kv_reduce_kernel(const float* __restrict__ part,
                                                        float* __restrict__ kv) {
  int idx = blockIdx.x * 256 + threadIdx.x;    // H*64*64 = 32768
  int h = idx >> 12, dm = idx & 4095;
  float s = 0.f;
  for (int sp = 0; sp < KV_SPLITS; ++sp) s += part[((size_t)h * KV_SPLITS + sp) * 4096 + dm];
  kv[idx] = s;
}

// ---------------------------------------------------------------------------
// o[l][h*64+m] = sink_alloc*sink_in * sum_d q[l][h,d]*kv[h][d][m]  -> bf16
// Block: 2 heads (hp), 256 l-rows; kv pair staged in LDS (32KB).
// ---------------------------------------------------------------------------
__global__ __launch_bounds__(256) void attn_o_kernel(const float* __restrict__ q,
                                                     const float* __restrict__ kv,
                                                     const float* __restrict__ sink_in,
                                                     const float* __restrict__ sink_alloc,
                                                     bf16_t* __restrict__ attn_ob) {
  const int lblk = blockIdx.x, hp = blockIdx.y, t = threadIdx.x;
  const int l0 = lblk * 256;
  __shared__ float kvs[2 * 64 * 64];
  __shared__ float qs[2][128];
  for (int i = t; i < 8192; i += 256) kvs[i] = kv[hp * 8192 + i];
  const int r = t >> 7, c = t & 127;
  const int hl = c >> 6, m = c & 63;
  const int h = hp * 2 + hl;
  const float* kvh = kvs + hl * 4096;
  for (int li = 0; li < 256; li += 2) {
    int l = l0 + li + r;
    __syncthreads();
    qs[r][c] = q[(size_t)l * D_MODEL + hp * 128 + c];
    __syncthreads();
    float s = 0.f;
#pragma unroll 8
    for (int d = 0; d < 64; ++d) s = fmaf(qs[r][hl * 64 + d], kvh[d * 64 + m], s);
    float val = s * sink_in[h * L_SEQ + l] * sink_alloc[h * L_SEQ + l];
    attn_ob[(size_t)l * D_MODEL + hp * 128 + c] = f2bf(val);
  }
}

// x2[b,l,d] = x[b,l,d] + sa[l,d]  (L*D = 2^23)
__global__ __launch_bounds__(256) void add_sa_kernel(const float* __restrict__ x,
                                                     const float* __restrict__ sa,
                                                     float* __restrict__ out, int n) {
  int i = blockIdx.x * 256 + threadIdx.x;
  if (i < n) out[i] = x[i] + sa[i & ((1 << 23) - 1)];
}

// ---------------------------------------------------------------------------
extern "C" void kernel_launch(void* const* d_in, const int* in_sizes, int n_in,
                              void* d_out, int out_size, void* d_ws, size_t ws_size,
                              hipStream_t stream) {
  const float* x   = (const float*)d_in[0];
  const float* Wq  = (const float*)d_in[1];
  const float* bq  = (const float*)d_in[2];
  const float* Wk  = (const float*)d_in[3];
  const float* bk  = (const float*)d_in[4];
  const float* Wv  = (const float*)d_in[5];
  const float* bv  = (const float*)d_in[6];
  const float* Wo  = (const float*)d_in[7];
  const float* bo  = (const float*)d_in[8];
  const float* W1  = (const float*)d_in[9];
  const float* b1  = (const float*)d_in[10];
  const float* W2  = (const float*)d_in[11];
  const float* b2  = (const float*)d_in[12];
  const float* g1  = (const float*)d_in[13];
  const float* be1 = (const float*)d_in[14];
  const float* g2  = (const float*)d_in[15];
  const float* be2 = (const float*)d_in[16];
  float* out = (float*)d_out;

  char* ws = (char*)d_ws;
  size_t off = 0;
  auto alloc = [&](size_t bytes) -> char* {
    char* p = ws + off;
    off += (bytes + 255) & ~(size_t)255;
    return p;
  };

  const size_t LD = (size_t)L_SEQ * D_MODEL;
  bf16_t* h1b   = (bf16_t*)alloc(LD * 2);
  bf16_t* wqb   = (bf16_t*)alloc((size_t)D_MODEL * D_MODEL * 2);
  bf16_t* wkb   = (bf16_t*)alloc((size_t)D_MODEL * D_MODEL * 2);
  bf16_t* wvb   = (bf16_t*)alloc((size_t)D_MODEL * D_MODEL * 2);
  bf16_t* wob   = (bf16_t*)alloc((size_t)D_MODEL * D_MODEL * 2);
  bf16_t* w1b   = (bf16_t*)alloc((size_t)D_MODEL * F_FFN * 2);
  bf16_t* w2b   = (bf16_t*)alloc((size_t)F_FFN * D_MODEL * 2);
  float*  qf    = (float*)alloc(LD * 4);
  float*  kf    = (float*)alloc(LD * 4);
  float*  vf    = (float*)alloc(LD * 4);
  float*  qsum  = (float*)alloc(D_MODEL * 4);
  float*  ksum  = (float*)alloc(D_MODEL * 4);
  float*  qwsum = (float*)alloc(D_MODEL * 4);
  float*  kwsum = (float*)alloc(D_MODEL * 4);
  float*  sink_in    = (float*)alloc((size_t)N_HEADS * L_SEQ * 4);
  float*  src_out    = (float*)alloc((size_t)N_HEADS * L_SEQ * 4);
  float*  cons_src   = (float*)alloc((size_t)N_HEADS * L_SEQ * 4);
  float*  sink_alloc = (float*)alloc((size_t)N_HEADS * L_SEQ * 4);
  float*  comp       = (float*)alloc((size_t)N_HEADS * L_SEQ * 4);
  float*  smpart     = (float*)alloc(N_HEADS * 16 * 4);
  float*  kvpart     = (float*)alloc((size_t)N_HEADS * KV_SPLITS * 64 * 64 * 4);
  float*  kvbuf      = (float*)alloc((size_t)N_HEADS * 64 * 64 * 4);
  bf16_t* attn_ob    = (bf16_t*)alloc(LD * 2);
  float*  sa         = (float*)alloc(LD * 4);
  bf16_t* h2b        = (bf16_t*)alloc(2 * LD * 2);
  bf16_t* midb       = (bf16_t*)alloc((size_t)2 * L_SEQ * F_FFN * 2);

  // 1. weights -> bf16
  cvt_bf16_kernel<<<(262144 + 255) / 256, 256, 0, stream>>>(Wq, wqb, 262144);
  cvt_bf16_kernel<<<(262144 + 255) / 256, 256, 0, stream>>>(Wk, wkb, 262144);
  cvt_bf16_kernel<<<(262144 + 255) / 256, 256, 0, stream>>>(Wv, wvb, 262144);
  cvt_bf16_kernel<<<(262144 + 255) / 256, 256, 0, stream>>>(Wo, wob, 262144);
  cvt_bf16_kernel<<<(1048576 + 255) / 256, 256, 0, stream>>>(W1, w1b, 1048576);
  cvt_bf16_kernel<<<(1048576 + 255) / 256, 256, 0, stream>>>(W2, w2b, 1048576);

  // 2. LN1 on batch 0 only (reference keeps attn(...)[0])
  layernorm_kernel<<<L_SEQ, 256, 0, stream>>>(x, g1, be1, h1b);

  // 3. Q/K/V projections (WMMA), sigmoid fused for Q,K
  dim3 gqkv(D_MODEL / BN, L_SEQ / BM);
  gemm_bf16_kernel<1><<<gqkv, 256, 0, stream>>>(h1b, wqb, bq, nullptr, qf, nullptr, L_SEQ, D_MODEL, D_MODEL);
  gemm_bf16_kernel<1><<<gqkv, 256, 0, stream>>>(h1b, wkb, bk, nullptr, kf, nullptr, L_SEQ, D_MODEL, D_MODEL);
  gemm_bf16_kernel<0><<<gqkv, 256, 0, stream>>>(h1b, wvb, bv, nullptr, vf, nullptr, L_SEQ, D_MODEL, D_MODEL);

  // 4. flow-conservation statistics
  colsum2_kernel<<<D_MODEL, 256, 0, stream>>>(qf, kf, qsum, ksum);
  flow_inv_kernel<<<512, 256, 0, stream>>>(qf, kf, ksum, qsum, sink_in, src_out);
  wcolsum_kernel<<<D_MODEL, 256, 0, stream>>>(qf, kf, sink_in, src_out, qwsum, kwsum);
  flow_cons_kernel<<<512, 256, 0, stream>>>(qf, kf, kwsum, qwsum, sink_alloc, cons_src);
  softmax_sum_kernel<<<N_HEADS * 16, 256, 0, stream>>>(cons_src, smpart);
  softmax_comp_kernel<<<(N_HEADS * L_SEQ) / 256, 256, 0, stream>>>(cons_src, smpart, comp);

  // 5. kv = k^T (v * comp), per head 64x64
  kv_partial_kernel<<<dim3(KV_SPLITS, N_HEADS), 256, 0, stream>>>(kf, vf, comp, kvpart);
  kv_reduce_kernel<<<(N_HEADS * 64 * 64) / 256, 256, 0, stream>>>(kvpart, kvbuf);

  // 6. o = (q*sink_in) kv * sink_alloc -> bf16 [L,D]
  attn_o_kernel<<<dim3(L_SEQ / 256, N_HEADS / 2), 256, 0, stream>>>(qf, kvbuf, sink_in, sink_alloc, attn_ob);

  // 7. output projection (WMMA) -> sa
  gemm_bf16_kernel<0><<<gqkv, 256, 0, stream>>>(attn_ob, wob, bo, nullptr, sa, nullptr, L_SEQ, D_MODEL, D_MODEL);

  // 8. x2 = x + sa (broadcast over batch) -> d_out
  add_sa_kernel<<<(2 * (int)LD) / 256, 256, 0, stream>>>(x, sa, out, 2 * (int)LD);

  // 9. LN2 over both batches
  layernorm_kernel<<<2 * L_SEQ, 256, 0, stream>>>(out, g2, be2, h2b);

  // 10. FFN GEMM1 with exact-GELU epilogue -> bf16 mid
  dim3 gffn1(F_FFN / BN, (2 * L_SEQ) / BM);
  gemm_bf16_kernel<2><<<gffn1, 256, 0, stream>>>(h2b, w1b, b1, nullptr, nullptr, midb, 2 * L_SEQ, F_FFN, D_MODEL);

  // 11. FFN GEMM2 + bias + residual(x2) -> d_out
  dim3 gffn2(D_MODEL / BN, (2 * L_SEQ) / BM);
  gemm_bf16_kernel<3><<<gffn2, 256, 0, stream>>>(midb, w2b, b2, out, out, nullptr, 2 * L_SEQ, D_MODEL, F_FFN);

  (void)in_sizes; (void)n_in; (void)out_size; (void)ws_size;
}
